// CustomMultiheadAttention_12395275616468
// MI455X (gfx1250) — matrix operand order repaired and checked by reading, not compiled, not run on hardware
//
#include <hip/hip_runtime.h>

// ---------------------------------------------------------------------------
// CDNA5 (gfx1250) fused multi-head attention pipeline, bf16 WMMA everywhere.
//   B=4, N=2048, C=1024, H=16, D=64
// Stages (all on `stream`):
//   1) cvt activations fp32->bf16          (q,k,v inputs)
//   2) cvt+transpose weights fp32->bf16    (Wq,Wk,Wv,Wp -> W^T)
//   3) GEMM (WMMA bf16, fp32 accum) x3     -> Qh[B,H,N,D], Kh[B,H,N,D],
//                                             Vt[B,H,D,N], bias fused
//   4) flash attention (WMMA bf16)         -> Y[B,N,C] bf16
//   5) GEMM (WMMA)                         -> out fp32 [B,N,C], bias fused
// LDS tiles are double-buffered and staged with GLOBAL_LOAD_ASYNC_TO_LDS_B128
// (ASYNCcnt path), overlapping HBM->LDS copies with the WMMA stream; one
// workgroup barrier per K-step.
// ---------------------------------------------------------------------------

typedef __attribute__((ext_vector_type(16))) __bf16 v16bf;
typedef __attribute__((ext_vector_type(8)))  float  v8f;

union V16 {
    v16bf v;
    uint4 q[2];
};

__device__ __forceinline__ unsigned short f32_to_bf16(float f) {
    unsigned int u = __float_as_uint(f);
    unsigned int r = u + 0x7FFFu + ((u >> 16) & 1u);   // round-to-nearest-even
    return (unsigned short)(r >> 16);
}

__device__ __forceinline__ v8f v8f_zero() {
    v8f z = {0.f, 0.f, 0.f, 0.f, 0.f, 0.f, 0.f, 0.f};
    return z;
}

__device__ __forceinline__ v8f wmma_bf16(const V16& a, const V16& b, v8f c) {
    // D = A(16x32 bf16) x B(32x16 bf16) + C(16x16 f32)
    return __builtin_amdgcn_wmma_f32_16x16x32_bf16(
        /*neg_a=*/false, a.v, /*neg_b=*/false, b.v,
        /*c_mod=*/(short)0, c, /*reuse_a=*/false, /*reuse_b=*/false);
}

// ---- async memory->LDS 16B copy (per active lane), tracked by ASYNCcnt ----
__device__ __forceinline__ void async_copy16(void* lds_ptr, const void* gptr) {
    // Flat pointers to __shared__ carry the wave-relative LDS byte offset in
    // the low 32 bits (ISA 10.2).
    unsigned ldsoff = (unsigned)(unsigned long long)(uintptr_t)lds_ptr;
    unsigned long long ga = (unsigned long long)(uintptr_t)gptr;
    asm volatile("global_load_async_to_lds_b128 %0, %1, off"
                 :: "v"(ldsoff), "v"(ga)
                 : "memory");
}

__device__ __forceinline__ void wait_async0() {
    asm volatile("s_wait_asynccnt 0x0" ::: "memory");
}

// Wave-local LDS fence: LDS ops of one wave complete in order; this both
// drains DScnt and stops the compiler reordering the P store->load pair.
__device__ __forceinline__ void wave_lds_fence() {
    asm volatile("s_wait_dscnt 0x0" ::: "memory");
}

// ---------------------------------------------------------------------------
// Problem constants
// ---------------------------------------------------------------------------
#define BSZ   4
#define NPOS  2048
#define CDIM  1024
#define HEADS 16
#define DHEAD 64
#define NROWS (BSZ * NPOS)          // 8192

// ---------------------------------------------------------------------------
// 1) elementwise fp32 -> bf16
// ---------------------------------------------------------------------------
__global__ void cvt_bf16_kernel(const float* __restrict__ in,
                                unsigned short* __restrict__ out, long n) {
    long i = (long)blockIdx.x * blockDim.x + threadIdx.x;
    long stride = (long)gridDim.x * blockDim.x;
    for (; i < n; i += stride) out[i] = f32_to_bf16(in[i]);
}

// ---------------------------------------------------------------------------
// 2) weight transpose + convert: Wt[n*C + k] = bf16(W[k*C + n])
// ---------------------------------------------------------------------------
__global__ void cvt_transpose_kernel(const float* __restrict__ W,
                                     unsigned short* __restrict__ Wt) {
    long o = (long)blockIdx.x * blockDim.x + threadIdx.x;   // output index
    int n = (int)(o >> 10);
    int k = (int)(o & (CDIM - 1));
    Wt[o] = f32_to_bf16(W[(long)k * CDIM + n]);
}

// ---------------------------------------------------------------------------
// 3/5) generic WMMA GEMM: out = A[8192,1024] @ Wt^T + bias
//   A   : bf16 row-major [NROWS, C]
//   Bt  : bf16 "W transposed" [C, C], Bt[n][k] = W[k][n]
//   mode 0: bf16 out, split-head  [B,H,N,D]
//   mode 1: bf16 out, transposed  [B,H,D,N]   (for V)
//   mode 2: fp32 out, plain       [row, col]  (final projection)
// Block: 256 thr (8 waves). Tile: 128 rows x 64 cols. K-step 32.
// Double-buffered async staging; 1 barrier per K-step.
// ---------------------------------------------------------------------------
__global__ __launch_bounds__(256) void gemm_bf16_kernel(
    const unsigned short* __restrict__ A,
    const unsigned short* __restrict__ Bt,
    const float* __restrict__ bias,
    unsigned short* __restrict__ out_b,
    float* __restrict__ out_f,
    int mode) {
    __shared__ __align__(16) unsigned short Xl[2][128 * 32];
    __shared__ __align__(16) unsigned short Wl[2][64 * 32];

    const int t    = threadIdx.x;
    const int wave = t >> 5;
    const int lane = t & 31;
    const int ln   = lane & 15;
    const int hi   = lane >> 4;

    const int row0 = blockIdx.y * 128;
    const int col0 = blockIdx.x * 64;

    // async-stage one K-step tile pair into buffer `buf`
    auto stage = [&](int kk, int buf) {
        const int r = t >> 1, half = t & 1;
        const unsigned short* srcA = A + (size_t)(row0 + r) * CDIM + kk + half * 16;
        unsigned short* dstA = &Xl[buf][r * 32 + half * 16];
        async_copy16(dstA, srcA);
        async_copy16(dstA + 8, srcA + 8);
        const int n = t >> 2, qq = t & 3;
        async_copy16(&Wl[buf][n * 32 + qq * 8],
                     Bt + (size_t)(col0 + n) * CDIM + kk + qq * 8);
    };

    v8f acc[4];
#pragma unroll
    for (int c = 0; c < 4; ++c) acc[c] = v8f_zero();

    stage(0, 0);
    wait_async0();
    __syncthreads();

    for (int kk = 0; kk < CDIM; kk += 32) {
        const int cur = (kk >> 5) & 1;
        // issue next tile's async copies; they overlap the WMMAs below
        if (kk + 32 < CDIM) stage(kk + 32, cur ^ 1);
        if (kk + 64 < CDIM)
            __builtin_prefetch(A + (size_t)(row0 + (t >> 1)) * CDIM + kk + 64, 0, 1);

        // ---- A-matrix regs (16x32, ISA layout: lane=row, kbase=(lane>>4)*8)
        V16 am;
        {
            const int arow = wave * 16 + ln;
            const int kb   = hi * 8;
            am.q[0] = *(const uint4*)(&Xl[cur][arow * 32 + kb]);
            am.q[1] = *(const uint4*)(&Xl[cur][arow * 32 + kb + 16]);
        }
        // ---- 4 column chunks of 16 ----
#pragma unroll
        for (int c = 0; c < 4; ++c) {
            V16 bm;
            const int bn  = c * 16 + ln;
            const int kb2 = hi * 8;     // lanes 0-15: K 0..7,16..23, lanes 16-31: K 8..15,24..31
            bm.q[0] = *(const uint4*)(&Wl[cur][bn * 32 + kb2]);
            bm.q[1] = *(const uint4*)(&Wl[cur][bn * 32 + kb2 + 16]);
            acc[c]  = wmma_bf16(am, bm, acc[c]);
        }

        // single barrier: (a) everyone finished reading buf `cur`
        //                 (b) everyone's copies into buf `cur^1` have landed
        wait_async0();
        __syncthreads();
    }

    // ---- epilogue: bias + store (C/D layout: vgpr r -> row r + 8*hi, lane -> col)
#pragma unroll
    for (int c = 0; c < 4; ++c) {
        const int col = col0 + c * 16 + ln;
        const float bv = bias[col];
#pragma unroll
        for (int r = 0; r < 8; ++r) {
            const int row = row0 + wave * 16 + hi * 8 + r;
            const float val = acc[c][r] + bv;
            if (mode == 2) {
                out_f[(size_t)row * CDIM + col] = val;
            } else {
                const int b_ = row >> 11;        // row / NPOS
                const int n_ = row & (NPOS - 1);
                const int h_ = col >> 6;         // col / DHEAD
                const int d_ = col & (DHEAD - 1);
                size_t addr;
                if (mode == 0)
                    addr = (((size_t)b_ * HEADS + h_) * NPOS + n_) * DHEAD + d_;
                else
                    addr = (((size_t)b_ * HEADS + h_) * DHEAD + d_) * NPOS + n_;
                out_b[addr] = f32_to_bf16(val);
            }
        }
    }
}

// ---------------------------------------------------------------------------
// 4) flash attention, one (b,h,qtile) per block, 8 waves x 16 query rows.
//    Qh,Kh: bf16 [B,H,N,D]; Vt: bf16 [B,H,D,N]; Y: bf16 [B,N,C].
//    Key tile = 64, online softmax with additive frequency bias.
//    Double-buffered async K/V staging; 1 barrier per key tile (the P
//    re-layout is wave-private LDS and needs only a wave-local DS fence).
// ---------------------------------------------------------------------------
__global__ __launch_bounds__(256) void flash_attn_kernel(
    const unsigned short* __restrict__ Qh,
    const unsigned short* __restrict__ Kh,
    const unsigned short* __restrict__ Vt,
    unsigned short* __restrict__ Y) {
    __shared__ __align__(16) unsigned short Kl[2][64 * 64];   // [key][d]
    __shared__ __align__(16) unsigned short Vl[2][64 * 64];   // [d][key]
    __shared__ __align__(16) unsigned short Pl[8][16 * 64];   // per-wave P tile

    const int t    = threadIdx.x;
    const int wave = t >> 5;
    const int lane = t & 31;
    const int ln   = lane & 15;
    const int hi   = lane >> 4;

    const int qt = blockIdx.x;
    const int h  = blockIdx.y;
    const int b  = blockIdx.z;

    const size_t headoff = ((size_t)b * HEADS + h) * ((size_t)NPOS * DHEAD);
    const unsigned short* Qp = Qh + headoff;   // [N][64]
    const unsigned short* Kp = Kh + headoff;   // [N][64]
    const unsigned short* Vp = Vt + headoff;   // [64][N]

    const int qrow0 = qt * 128 + wave * 16;

    // async-stage one key-tile of K and V^T into buffer `buf`
    auto stageKV = [&](int kt, int buf) {
        const unsigned short* srcK = Kp + (size_t)kt * DHEAD;  // 8KB contiguous
        async_copy16(&Kl[buf][t * 8], srcK + t * 8);
        async_copy16(&Kl[buf][(t + 256) * 8], srcK + (t + 256) * 8);
        const int d = t >> 2, qq = t & 3;
        const unsigned short* srcV = Vp + (size_t)d * NPOS + kt + qq * 16;
        unsigned short* dstV = &Vl[buf][d * 64 + qq * 16];
        async_copy16(dstV, srcV);
        async_copy16(dstV + 8, srcV + 8);
    };

    // ---- Q in A-layout registers (2 chunks over d: 0..31, 32..63) ----
    V16 qa[2];
    {
        const size_t r  = (size_t)(qrow0 + ln) * DHEAD;
        const int    kb = hi * 8;
#pragma unroll
        for (int a2 = 0; a2 < 2; ++a2) {
            qa[a2].q[0] = *(const uint4*)(Qp + r + a2 * 32 + kb);
            qa[a2].q[1] = *(const uint4*)(Qp + r + a2 * 32 + kb + 16);
        }
    }

    v8f o[4];
#pragma unroll
    for (int c = 0; c < 4; ++c) o[c] = v8f_zero();
    float m[8], l[8];
#pragma unroll
    for (int r = 0; r < 8; ++r) { m[r] = -1e30f; l[r] = 0.f; }

    const float scale = 0.125f;   // 1/sqrt(64)

    stageKV(0, 0);
    wait_async0();
    __syncthreads();

    for (int kt = 0; kt < NPOS; kt += 64) {
        const int cur = (kt >> 6) & 1;
        // issue next key-tile copies; overlap with S/softmax/PV below
        if (kt + 64 < NPOS) stageKV(kt + 64, cur ^ 1);

        // ---- S = Q K^T (unscaled) : 4 chunks of 16 keys, K-dim = 64 (2 WMMAs)
        v8f s[4];
#pragma unroll
        for (int c = 0; c < 4; ++c) {
            s[c] = v8f_zero();
            const int key = c * 16 + ln;
            const int kb2 = hi * 8;
#pragma unroll
            for (int a2 = 0; a2 < 2; ++a2) {
                V16 bm;
                bm.q[0] = *(const uint4*)(&Kl[cur][key * 64 + a2 * 32 + kb2]);
                bm.q[1] = *(const uint4*)(&Kl[cur][key * 64 + a2 * 32 + kb2 + 16]);
                s[c]    = wmma_bf16(qa[a2], bm, s[c]);
            }
        }

        // ---- frequency bias per key column ----
        float fb[4];
#pragma unroll
        for (int c = 0; c < 4; ++c) {
            const int kg = kt + c * 16 + ln;
            const float fr = (float)kg * (1.0f / (float)(NPOS - 1));
            const float d0 = fr - 0.5f;
            fb[c] = -d0 * d0 * 10.0f;
        }

        // ---- online softmax (rows striped over 16-lane halves) ----
        float p[4][8];
#pragma unroll
        for (int r = 0; r < 8; ++r) {
            float mx = -1e30f;
#pragma unroll
            for (int c = 0; c < 4; ++c) {
                const float v = s[c][r] * scale + fb[c];
                p[c][r] = v;
                mx = fmaxf(mx, v);
            }
#pragma unroll
            for (int off = 8; off >= 1; off >>= 1)
                mx = fmaxf(mx, __shfl_xor(mx, off, 32));
            const float newm = fmaxf(m[r], mx);
            const float corr = __expf(m[r] - newm);
            float sum = 0.f;
#pragma unroll
            for (int c = 0; c < 4; ++c) {
                const float e = __expf(p[c][r] - newm);
                p[c][r] = e;
                sum += e;
            }
#pragma unroll
            for (int off = 8; off >= 1; off >>= 1)
                sum += __shfl_xor(sum, off, 32);
            l[r] = l[r] * corr + sum;
            m[r] = newm;
#pragma unroll
            for (int c = 0; c < 4; ++c) o[c][r] *= corr;
        }

        // ---- write P (C-layout regs -> row-major LDS tile 16x64) ----
        // Wave-private region: LDS ops within a wave complete in order, so a
        // wave-local DS fence suffices (no workgroup barrier).
        unsigned short* Pw = &Pl[wave][0];
#pragma unroll
        for (int r = 0; r < 8; ++r)
#pragma unroll
            for (int c = 0; c < 4; ++c)
                Pw[(hi * 8 + r) * 64 + c * 16 + ln] = f32_to_bf16(p[c][r]);
        wave_lds_fence();

        // ---- O += P V : A = P (A-layout from LDS), B = V (from Vt tile) ----
        const int kb = hi * 8;
#pragma unroll
        for (int a2 = 0; a2 < 2; ++a2) {
            V16 pa;
            pa.q[0] = *(const uint4*)(Pw + ln * 64 + a2 * 32 + kb);
            pa.q[1] = *(const uint4*)(Pw + ln * 64 + a2 * 32 + kb + 16);
#pragma unroll
            for (int c = 0; c < 4; ++c) {
                V16 bm;
                const int dcol = c * 16 + ln;
                const int kb2  = hi * 8;
                bm.q[0] = *(const uint4*)(&Vl[cur][dcol * 64 + a2 * 32 + kb2]);
                bm.q[1] = *(const uint4*)(&Vl[cur][dcol * 64 + a2 * 32 + kb2 + 16]);
                o[c]    = wmma_bf16(pa, bm, o[c]);
            }
        }

        // single barrier per tile: everyone done reading buf `cur`, and
        // everyone's async copies into buf `cur^1` have landed
        wait_async0();
        __syncthreads();
    }

    // ---- normalize + store Y bf16 [B,N,C] ----
#pragma unroll
    for (int c = 0; c < 4; ++c) {
#pragma unroll
        for (int r = 0; r < 8; ++r) {
            const int row = qrow0 + hi * 8 + r;                // position n
            const int col = h * DHEAD + c * 16 + ln;           // channel
            const float val = o[c][r] / l[r];
            Y[((size_t)b * NPOS + row) * CDIM + col] = f32_to_bf16(val);
        }
    }
}

// ---------------------------------------------------------------------------
// Host launch
// ---------------------------------------------------------------------------
extern "C" void kernel_launch(void* const* d_in, const int* in_sizes, int n_in,
                              void* d_out, int out_size, void* d_ws, size_t ws_size,
                              hipStream_t stream) {
    const float* q_in = (const float*)d_in[0];
    const float* k_in = (const float*)d_in[1];
    const float* v_in = (const float*)d_in[2];
    const float* Wq = (const float*)d_in[3];
    const float* bq = (const float*)d_in[4];
    const float* Wk = (const float*)d_in[5];
    const float* bk = (const float*)d_in[6];
    const float* Wv = (const float*)d_in[7];
    const float* bv = (const float*)d_in[8];
    const float* Wp = (const float*)d_in[9];
    const float* bp = (const float*)d_in[10];
    float* out = (float*)d_out;

    // Workspace layout (bf16 elements). Total ~126 MB.
    const size_t NX = (size_t)NROWS * CDIM;   // 8,388,608
    const size_t NW = (size_t)CDIM * CDIM;    // 1,048,576
    unsigned short* ws  = (unsigned short*)d_ws;
    unsigned short* qx  = ws;
    unsigned short* kx  = qx + NX;
    unsigned short* vx  = kx + NX;
    unsigned short* wqt = vx + NX;
    unsigned short* wkt = wqt + NW;
    unsigned short* wvt = wkt + NW;
    unsigned short* wpt = wvt + NW;
    unsigned short* Qh  = wpt + NW;
    unsigned short* Kh  = Qh + NX;
    unsigned short* Vt  = Kh + NX;
    unsigned short* Yb  = Vt + NX;

    // 1) activations -> bf16
    cvt_bf16_kernel<<<2048, 256, 0, stream>>>(q_in, qx, (long)NX);
    cvt_bf16_kernel<<<2048, 256, 0, stream>>>(k_in, kx, (long)NX);
    cvt_bf16_kernel<<<2048, 256, 0, stream>>>(v_in, vx, (long)NX);

    // 2) weights -> bf16 transposed
    const int wblocks = (int)(NW / 256);
    cvt_transpose_kernel<<<wblocks, 256, 0, stream>>>(Wq, wqt);
    cvt_transpose_kernel<<<wblocks, 256, 0, stream>>>(Wk, wkt);
    cvt_transpose_kernel<<<wblocks, 256, 0, stream>>>(Wv, wvt);
    cvt_transpose_kernel<<<wblocks, 256, 0, stream>>>(Wp, wpt);

    // 3) QKV projections (WMMA)
    dim3 gg(CDIM / 64, NROWS / 128);   // (16, 64)
    gemm_bf16_kernel<<<gg, 256, 0, stream>>>(qx, wqt, bq, Qh, nullptr, 0);
    gemm_bf16_kernel<<<gg, 256, 0, stream>>>(kx, wkt, bk, Kh, nullptr, 0);
    gemm_bf16_kernel<<<gg, 256, 0, stream>>>(vx, wvt, bv, Vt, nullptr, 1);

    // 4) flash attention
    dim3 ga(NPOS / 128, HEADS, BSZ);   // (16, 16, 4)
    flash_attn_kernel<<<ga, 256, 0, stream>>>(Qh, Kh, Vt, Yb);

    // 5) output projection -> fp32 d_out
    gemm_bf16_kernel<<<gg, 256, 0, stream>>>(Yb, wpt, bp, nullptr, out, 2);
}
